// FastAttentionBottleneck_30356828848814
// MI455X (gfx1250) — matrix operand order repaired and checked by reading, not compiled, not run on hardware
//
#include <hip/hip_runtime.h>
#include <hip/hip_bf16.h>
#include <math.h>

typedef __attribute__((ext_vector_type(16))) _Float16 v16h;
typedef __attribute__((ext_vector_type(8)))  float    v8f;

#define B_     16
#define C_     512
#define L_     4096
#define D_     384
#define NW_    256      // local window count (tokens)
#define INNER_ 512
#define H_     8
#define DH_    64
#define FF_    1536
#define ROWS_  (B_ * NW_)   // 4096 flattened (b, n) rows

__device__ __forceinline__ float gelu_exact(float x) {
    return 0.5f * x * (1.0f + erff(x * 0.7071067811865475f));
}

// Load one 16x32 f16 A fragment: lane M=l15 holds K chunk
// [k0+16*hi, +16). Arow is lane-fixed (row base + 16*hi).
__device__ __forceinline__ v16h load_afrag(const float* __restrict__ Arow,
                                           int k0, bool valid) {
    v16h a;
    if (valid) {
        const float* p0 = Arow + k0;
        const float* p1 = Arow + k0 + 8;
        #pragma unroll
        for (int i = 0; i < 8; i++) {
            a[i]     = (_Float16)p0[i];
            a[8 + i] = (_Float16)p1[i];
        }
    } else {
        #pragma unroll
        for (int i = 0; i < 16; i++) a[i] = (_Float16)0.0f;
    }
    return a;
}

// ---------------------------------------------------------------------------
// Generic WMMA f16 GEMM:  Out[M,N] (+)= A[M,K] @ W + bias, optional GELU.
// All dims/strides are compile-time template params so global loads use
// immediate offsets and the K-loop unrolls into clausable load groups.
//   WT=false: W is [K,N] row-major.   WT=true: W is [N,K] row-major.
//   SHIFT: conv mode — A row r reads row r+shift; rows with (r&255)+shift
//          outside [0,256) contribute zero (padding=1 Conv1d over n).
// Block: 128 threads = 4 waves, block tile 64(M) x 64(N).
// Wave (wm = w&1, wn = w>>1) computes a 32x32 tile = 2 A frags x 2 B frags
// -> 4 WMMAs per K-step of 32 (halves fragment loads per WMMA vs 16x32).
// grid = (N/64, M/64, Z); per-z strides for batched (per-head) GEMMs.
// ---------------------------------------------------------------------------
template<int N, int K, int LDA, int LDO,
         bool WT, bool BIAS, bool ACCUM, bool GELU_E, bool SHIFT>
__global__ __launch_bounds__(128) void gemm_wmma(
    const float* __restrict__ A, long long sAz,
    const float* __restrict__ W, long long sWz,
    const float* __restrict__ bias,
    float* __restrict__ Out, long long sOz,
    int shift)
{
    int z = blockIdx.z;
    A   += z * sAz;
    W   += z * sWz;
    Out += z * sOz;

    const int m0   = blockIdx.y * 64;
    const int n0   = blockIdx.x * 64;
    const int t    = threadIdx.x;
    const int wave = t >> 5;
    const int lane = t & 31;
    const int wm   = wave & 1;       // 32-row strip
    const int wn   = wave >> 1;      // 32-col strip
    const int hi   = lane >> 4;
    const int l15  = lane & 15;

    const int mrow0 = m0 + wm * 32 + l15;     // A rows for fragment 0 / 1
    const int mrow1 = mrow0 + 16;
    int ar0 = mrow0 * LDA;
    int ar1 = mrow1 * LDA;
    bool valid0 = true, valid1 = true;
    if (SHIFT) {
        int nl0 = (mrow0 & (NW_ - 1)) + shift;
        int nl1 = (mrow1 & (NW_ - 1)) + shift;
        valid0 = (nl0 >= 0) && (nl0 < NW_);
        valid1 = (nl1 >= 0) && (nl1 < NW_);
        ar0 += shift * LDA;
        ar1 += shift * LDA;
    }
    const float* Arow0 = A + ar0 + 16 * hi;   // lane-fixed bases
    const float* Arow1 = A + ar1 + 16 * hi;

    const int ncol0 = n0 + wn * 32 + l15;     // B/D col, first 16-tile
    const int ncol1 = ncol0 + 16;             // second 16-tile

    // lane-fixed W bases; in-loop offsets are compile-time immediates
    const float* Wb0;
    const float* Wb1;
    if (WT) {
        Wb0 = W + ncol0 * K + 16 * hi;
        Wb1 = W + ncol1 * K + 16 * hi;
    } else {
        Wb0 = W + 16 * hi * N + ncol0;
        Wb1 = W + 16 * hi * N + ncol1;
    }

    v8f acc00 = {}, acc01 = {}, acc10 = {}, acc11 = {};

    for (int k0 = 0; k0 < K; k0 += 32) {
        v16h a0 = load_afrag(Arow0, k0, !SHIFT || valid0);
        v16h a1 = load_afrag(Arow1, k0, !SHIFT || valid1);

        // B fragments (32x16 f16 each): lane N=l15,
        // lanes 0-15 hold K=k0..k0+15, lanes 16-31 hold K=k0+16..k0+31
        v16h b0, b1;
        if (WT) {
            const float* q0 = Wb0 + k0;
            const float* q1 = Wb1 + k0;
            #pragma unroll
            for (int i = 0; i < 16; i++) {
                b0[i] = (_Float16)q0[i];
                b1[i] = (_Float16)q1[i];
            }
        } else {
            const float* q0 = Wb0 + k0 * N;
            const float* q1 = Wb1 + k0 * N;
            #pragma unroll
            for (int i = 0; i < 16; i++) {
                b0[i] = (_Float16)q0[i * N];   // immediate offsets (i*N const)
                b1[i] = (_Float16)q1[i * N];
            }
        }
        acc00 = __builtin_amdgcn_wmma_f32_16x16x32_f16(
            false, a0, false, b0, (short)0, acc00, false, false);
        acc01 = __builtin_amdgcn_wmma_f32_16x16x32_f16(
            false, a0, false, b1, (short)0, acc01, false, false);
        acc10 = __builtin_amdgcn_wmma_f32_16x16x32_f16(
            false, a1, false, b0, (short)0, acc10, false, false);
        acc11 = __builtin_amdgcn_wmma_f32_16x16x32_f16(
            false, a1, false, b1, (short)0, acc11, false, false);
    }

    // ---- epilogue: C/D layout: VGPR v -> row (v + 8*hi), col = l15
    float bias0 = 0.f, bias1 = 0.f;
    if (BIAS) { bias0 = bias[ncol0]; bias1 = bias[ncol1]; }
    #pragma unroll
    for (int v = 0; v < 8; v++) {
        int r0 = m0 + wm * 32 + v + 8 * hi;   // rows of tile pair 0
        int r1 = r0 + 16;                     // rows of tile pair 1
        float x00 = acc00[v], x01 = acc01[v], x10 = acc10[v], x11 = acc11[v];
        if (BIAS) { x00 += bias0; x01 += bias1; x10 += bias0; x11 += bias1; }
        if (GELU_E) {
            x00 = gelu_exact(x00); x01 = gelu_exact(x01);
            x10 = gelu_exact(x10); x11 = gelu_exact(x11);
        }
        float* o0 = Out + r0 * LDO;
        float* o1 = Out + r1 * LDO;
        if (ACCUM) {
            o0[ncol0] += x00; o0[ncol1] += x01;
            o1[ncol0] += x10; o1[ncol1] += x11;
        } else {
            o0[ncol0] = x00; o0[ncol1] = x01;
            o1[ncol0] = x10; o1[ncol1] = x11;
        }
    }
}

// ---------------------------------------------------------------------------
// Adaptive avg-pool 16:1 and transpose: x[B,C,L] -> xpt[(b*256+w)*512 + c]
// ---------------------------------------------------------------------------
__global__ __launch_bounds__(256) void pool_k(
    const float* __restrict__ x, float* __restrict__ xpt)
{
    int tid = blockIdx.x * 256 + threadIdx.x;    // B*C*NW = 2,097,152
    int w  = tid & (NW_ - 1);
    int bc = tid >> 8;                           // b*C + c
    int c  = bc & (C_ - 1);
    int b  = bc >> 9;
    const float* p = x + ((long long)bc << 12) + w * 16;
    float s = 0.f;
    #pragma unroll
    for (int i = 0; i < 16; i++) s += p[i];
    xpt[((long long)(b * NW_ + w)) * C_ + c] = s * (1.0f / 16.0f);
}

// ---------------------------------------------------------------------------
// LayerNorm over D=384 (one block of 384 threads per row), optional fused add.
// Safe in-place (all reads happen before any write, enforced by barriers).
// ---------------------------------------------------------------------------
__global__ __launch_bounds__(384) void ln384(
    const float* __restrict__ in, const float* __restrict__ addsrc,
    const float* __restrict__ g, const float* __restrict__ b,
    float* __restrict__ out)
{
    __shared__ float ssum[12], ssq[12];
    int r = blockIdx.x;
    int t = threadIdx.x;
    long long idx = (long long)r * D_ + t;
    float x = in[idx];
    if (addsrc) x += addsrc[idx];
    float s = x, q = x * x;
    #pragma unroll
    for (int off = 16; off > 0; off >>= 1) {
        s += __shfl_xor(s, off, 32);
        q += __shfl_xor(q, off, 32);
    }
    int w = t >> 5;
    if ((t & 31) == 0) { ssum[w] = s; ssq[w] = q; }
    __syncthreads();
    if (t == 0) {
        float S = 0.f, Q = 0.f;
        for (int i = 0; i < 12; i++) { S += ssum[i]; Q += ssq[i]; }
        ssum[0] = S; ssq[0] = Q;
    }
    __syncthreads();
    float mean = ssum[0] * (1.0f / (float)D_);
    float var  = ssq[0] * (1.0f / (float)D_) - mean * mean;
    float inv  = rsqrtf(var + 1e-5f);
    out[idx] = (x - mean) * inv * g[t] + b[t];
}

// ---------------------------------------------------------------------------
// qkv [4096,1536] -> qh/kh/vh as [B*H, n, DH] with relu(q)*scale, relu(k).
// ---------------------------------------------------------------------------
__global__ __launch_bounds__(256) void split_heads_k(
    const float* __restrict__ qkv, float* __restrict__ qh,
    float* __restrict__ kh, float* __restrict__ vh)
{
    int tid = blockIdx.x * 256 + threadIdx.x;    // ROWS_*INNER_ = 2,097,152
    int c  = tid & (INNER_ - 1);
    int r  = tid >> 9;
    int d  = c & (DH_ - 1);
    int hh = c >> 6;
    int n  = r & (NW_ - 1);
    int b  = r >> 8;
    long long dst = (((long long)(b * H_ + hh) * NW_) + n) * DH_ + d;
    const float* src = qkv + (long long)r * (3 * INNER_);
    float qv = src[c];
    float kv = src[INNER_ + c];
    float vv = src[2 * INNER_ + c];
    qh[dst] = fmaxf(qv, 0.f) * 0.125f;   // SCALE = 64^-0.5
    kh[dst] = fmaxf(kv, 0.f);
    vh[dst] = vv;
}

// oh [B*H, n, DH] -> o [(b*n), H*DH]
__global__ __launch_bounds__(256) void merge_heads_k(
    const float* __restrict__ oh, float* __restrict__ o)
{
    int tid = blockIdx.x * 256 + threadIdx.x;
    int c  = tid & (INNER_ - 1);
    int r  = tid >> 9;
    int d  = c & (DH_ - 1);
    int hh = c >> 6;
    int n  = r & (NW_ - 1);
    int b  = r >> 8;
    o[tid] = oh[(((long long)(b * H_ + hh) * NW_) + n) * DH_ + d];
}

// local_w[D,D,3] -> wp[j][c*384 + d] = local_w[d][c][j]  (K-major for GEMM)
__global__ __launch_bounds__(256) void wpack_k(
    const float* __restrict__ lw, float* __restrict__ wp)
{
    int tid = blockIdx.x * 256 + threadIdx.x;    // 3*384*384 = 442,368
    int d = tid % D_;
    int c = (tid / D_) % D_;
    int j = tid / (D_ * D_);
    wp[j * D_ * D_ + c * D_ + d] = lw[(d * D_ + c) * 3 + j];
}

// out[b,c,l] = x[b,c,l] + lerp of projected small grid sb[(b*256+i)*512+c]
__global__ __launch_bounds__(256) void final_k(
    const float* __restrict__ x, const float* __restrict__ sb,
    float* __restrict__ out)
{
    int tid = blockIdx.x * 256 + threadIdx.x;    // B*C*L = 33,554,432
    int l  = tid & (L_ - 1);
    int bc = tid >> 12;
    int c  = bc & (C_ - 1);
    int b  = bc >> 9;
    float src = fmaxf((l + 0.5f) * (1.0f / 16.0f) - 0.5f, 0.0f);
    int i0 = (int)floorf(src);
    if (i0 > NW_ - 1) i0 = NW_ - 1;
    int i1 = i0 + 1;
    if (i1 > NW_ - 1) i1 = NW_ - 1;
    float w1 = src - (float)i0;
    float w0 = 1.0f - w1;
    long long base = (long long)b * NW_;
    float v = w0 * sb[(base + i0) * C_ + c] + w1 * sb[(base + i1) * C_ + c];
    out[tid] = v + x[tid];
}

// ---------------------------------------------------------------------------
extern "C" void kernel_launch(void* const* d_in, const int* in_sizes, int n_in,
                              void* d_out, int out_size, void* d_ws, size_t ws_size,
                              hipStream_t stream)
{
    (void)in_sizes; (void)n_in; (void)out_size; (void)ws_size;
    const float* x     = (const float*)d_in[0];
    const float* piw   = (const float*)d_in[1];
    const float* pib   = (const float*)d_in[2];
    const float* nin_g = (const float*)d_in[3];
    const float* nin_b = (const float*)d_in[4];
    const float* ln1_g = (const float*)d_in[5];
    const float* ln1_b = (const float*)d_in[6];
    const float* wqkv  = (const float*)d_in[7];
    const float* rf    = (const float*)d_in[8];
    const float* wout  = (const float*)d_in[9];
    const float* bout  = (const float*)d_in[10];
    const float* ln2_g = (const float*)d_in[11];
    const float* ln2_b = (const float*)d_in[12];
    const float* ffw1  = (const float*)d_in[13];
    const float* ffb1  = (const float*)d_in[14];
    const float* ffw2  = (const float*)d_in[15];
    const float* ffb2  = (const float*)d_in[16];
    const float* lw    = (const float*)d_in[17];
    const float* lb    = (const float*)d_in[18];
    const float* non_g = (const float*)d_in[19];
    const float* non_b = (const float*)d_in[20];
    const float* pow_  = (const float*)d_in[21];
    const float* pob   = (const float*)d_in[22];
    float* out = (float*)d_out;
    float* ws  = (float*)d_ws;

    // scratch layout (floats); total ~30.4M floats (~122 MB), L2-resident
    float* xpt = ws;                   // [4096, 512]        2,097,152
    float* h   = xpt + 2097152;        // [4096, 384]        1,572,864
    float* y   = h   + 1572864;        // [4096, 384]        1,572,864
    float* qkv = y   + 1572864;        // [4096, 1536]       6,291,456 (alias: g, sbuf)
    float* qh  = qkv + 6291456;        // [128, 256, 64]     2,097,152 (alias: obuf)
    float* kh  = qh  + 2097152;        //                    2,097,152 (alias: oh)
    float* vh  = kh  + 2097152;        //                    2,097,152
    float* qp  = vh  + 2097152;        //                    2,097,152
    float* kp  = qp  + 2097152;        //                    2,097,152
    float* att = kp  + 2097152;        // [128, 256, 256]    8,388,608 (alias: wp, tmp)
    float* wp   = att;                 // 442,368
    float* tmp  = att + 524288;        // 1,572,864
    float* obuf = qh;
    float* oh   = kh;
    float* g    = qkv;
    float* sb   = qkv;

    dim3 blk(256);
    dim3 gblk(128);   // GEMM blocks: 4 waves, 64x64 tile

    // pool + transpose, then proj_in (WT GEMM, bias), then LN (in-place)
    pool_k<<<ROWS_ * C_ / 256, blk, 0, stream>>>(x, xpt);
    gemm_wmma<D_, C_, C_, D_, true, true, false, false, false>
        <<<dim3(D_ / 64, ROWS_ / 64, 1), gblk, 0, stream>>>(
        xpt, 0, piw, 0, pib, h, 0, 0);
    ln384<<<ROWS_, D_, 0, stream>>>(h, nullptr, nin_g, nin_b, h);

    for (int i = 0; i < 2; i++) {
        ln384<<<ROWS_, D_, 0, stream>>>(h, nullptr, ln1_g + i * D_, ln1_b + i * D_, y);
        gemm_wmma<3 * INNER_, D_, D_, 3 * INNER_, false, false, false, false, false>
            <<<dim3(3 * INNER_ / 64, ROWS_ / 64, 1), gblk, 0, stream>>>(
            y, 0, wqkv + (long long)i * D_ * 3 * INNER_, 0, nullptr, qkv, 0, 0);
        split_heads_k<<<ROWS_ * INNER_ / 256, blk, 0, stream>>>(qkv, qh, kh, vh);
        // qp = relu(q)*s @ rf ; kp = relu(k) @ rf   (z = B*H = 128, rf shared)
        gemm_wmma<DH_, DH_, DH_, DH_, false, false, false, false, false>
            <<<dim3(1, 4, 128), gblk, 0, stream>>>(
            qh, NW_ * DH_, rf + i * DH_ * DH_, 0, nullptr, qp, NW_ * DH_, 0);
        gemm_wmma<DH_, DH_, DH_, DH_, false, false, false, false, false>
            <<<dim3(1, 4, 128), gblk, 0, stream>>>(
            kh, NW_ * DH_, rf + i * DH_ * DH_, 0, nullptr, kp, NW_ * DH_, 0);
        // attn = qp @ kp^T  (kp as [N,K] "weight" per z)
        gemm_wmma<NW_, DH_, DH_, NW_, true, false, false, false, false>
            <<<dim3(4, 4, 128), gblk, 0, stream>>>(
            qp, NW_ * DH_, kp, NW_ * DH_, nullptr, att, NW_ * NW_, 0);
        // oh = attn @ v
        gemm_wmma<DH_, NW_, NW_, DH_, false, false, false, false, false>
            <<<dim3(1, 4, 128), gblk, 0, stream>>>(
            att, NW_ * NW_, vh, NW_ * DH_, nullptr, oh, NW_ * DH_, 0);
        merge_heads_k<<<ROWS_ * INNER_ / 256, blk, 0, stream>>>(oh, obuf);
        // h += o @ wout + bout
        gemm_wmma<D_, INNER_, INNER_, D_, false, true, true, false, false>
            <<<dim3(D_ / 64, ROWS_ / 64, 1), gblk, 0, stream>>>(
            obuf, 0, wout + (long long)i * INNER_ * D_, 0, bout + i * D_, h, 0, 0);
        // FFN
        ln384<<<ROWS_, D_, 0, stream>>>(h, nullptr, ln2_g + i * D_, ln2_b + i * D_, y);
        gemm_wmma<FF_, D_, D_, FF_, false, true, false, true, false>
            <<<dim3(FF_ / 64, ROWS_ / 64, 1), gblk, 0, stream>>>(
            y, 0, ffw1 + (long long)i * D_ * FF_, 0, ffb1 + i * FF_, g, 0, 0);
        gemm_wmma<D_, FF_, FF_, D_, false, true, true, false, false>
            <<<dim3(D_ / 64, ROWS_ / 64, 1), gblk, 0, stream>>>(
            g, 0, ffw2 + (long long)i * FF_ * D_, 0, ffb2 + i * D_, h, 0, 0);
    }

    // local_context Conv1d(D,D,3,pad=1) over n as 3 shifted WMMA GEMMs
    wpack_k<<<3 * D_ * D_ / 256, blk, 0, stream>>>(lw, wp);
    gemm_wmma<D_, D_, D_, D_, false, true, false, false, true>
        <<<dim3(D_ / 64, ROWS_ / 64, 1), gblk, 0, stream>>>(
        h, 0, wp, 0, lb, tmp, 0, -1);
    gemm_wmma<D_, D_, D_, D_, false, false, true, false, true>
        <<<dim3(D_ / 64, ROWS_ / 64, 1), gblk, 0, stream>>>(
        h, 0, wp + D_ * D_, 0, nullptr, tmp, 0, 0);
    gemm_wmma<D_, D_, D_, D_, false, false, true, false, true>
        <<<dim3(D_ / 64, ROWS_ / 64, 1), gblk, 0, stream>>>(
        h, 0, wp + 2 * D_ * D_, 0, nullptr, tmp, 0, 1);

    // y = LN(h + conv), then proj_out on the SMALL grid (commutes with lerp)
    ln384<<<ROWS_, D_, 0, stream>>>(h, tmp, non_g, non_b, y);
    gemm_wmma<C_, D_, D_, C_, true, true, false, false, false>
        <<<dim3(C_ / 64, ROWS_ / 64, 1), gblk, 0, stream>>>(
        y, 0, pow_, 0, pob, sb, 0, 0);

    // upsample 256 -> 4096 (align_corners=False) + residual x
    final_k<<<(long long)(B_) * C_ * L_ / 256, blk, 0, stream>>>(x, sb, out);
}
